// GraphTransformer_11398843203994
// MI455X (gfx1250) — matrix operand order repaired and checked by reading, not compiled, not run on hardware
//
#include <hip/hip_runtime.h>
#include <hip/hip_bf16.h>

// ---------------------------------------------------------------------------
// Graph transformer forward for MI455X (gfx1250, wave32, WMMA).
//
// Design (compile-only; reasoned from MI455X specs):
//  * Edge tensors (ea: Etot x 64, ee: Etot x 128) stored bf16 -> halves the
//    ~2 GB/layer of HBM edge traffic that bounds this kernel (23.3 TB/s).
//  * All dense GEMMs use v_wmma_f32_16x16x32_bf16 (bf16 in, f32 accum),
//    block = 256 threads = 8 wave32 waves, 64x64 C tile per block,
//    16x32 C strip per wave (two 16x16 WMMA accumulators).
//  * B (weights, K x 64 slice <= 32 KB) is transpose-staged in LDS ONCE per
//    block (padded rows -> conflict-free ds_load_b128 fragment reads, two
//    uint4 loads per fragment); the K loop then runs with zero barriers and
//    zero divergence (A row index clamped to M-1; OOB rows never stored).
//    A fragments come straight from global (two 16B loads/lane, 64B-line
//    coalesced per lane pair) + global_prefetch of the next K line.
//  * Irregular segment ops (graph-LN, message scatter, softmax) use fp32
//    atomics; node accumulators (~13-26 MB) live in the 192 MB L2.
//  * Segment-max via order-preserving float->u32 key + atomicMax.
// ---------------------------------------------------------------------------

typedef __bf16 v16bf __attribute__((ext_vector_type(16)));
typedef float  v8f   __attribute__((ext_vector_type(8)));
typedef __bf16 bf16;

static constexpr int       EMB   = 64;
static constexpr int       HEADS = 2;
static constexpr int       LAYERS= 3;
static constexpr long long N_    = 50000;
static constexpr long long E_    = 1000000;
static constexpr long long G_    = 1024;
static constexpr long long NE_   = 200000;
static constexpr long long NTOT  = N_ + G_;                 // 51024
static constexpr long long ETOT  = E_ + 2*N_ + NTOT;        // 1151024
static constexpr float     SCALE = 0.125f;                  // 1/sqrt(64)

static inline int cdiv(long long a, long long b) { return (int)((a + b - 1) / b); }

// ------------------------------- utility kernels ---------------------------

__global__ void k_fill_f32(float* p, float v, long long n) {
  long long i = (long long)blockIdx.x * 256 + threadIdx.x;
  if (i < n) p[i] = v;
}
__global__ void k_fill_u32(unsigned* p, unsigned v, long long n) {
  long long i = (long long)blockIdx.x * 256 + threadIdx.x;
  if (i < n) p[i] = v;
}
__global__ void k_cvt_bf16(const float* __restrict__ s, bf16* __restrict__ d, long long n) {
  long long i = (long long)blockIdx.x * 256 + threadIdx.x;
  if (i < n) d[i] = (bf16)s[i];
}
__global__ void k_add_cvt(const float* __restrict__ a, const float* __restrict__ b,
                          bf16* __restrict__ o, long long n) {
  long long i = (long long)blockIdx.x * 256 + threadIdx.x;
  if (i < n) o[i] = (bf16)(a[i] + b[i]);
}

// ------------------------------- edge setup --------------------------------

__global__ void k_build_edges(const int* __restrict__ edge_index,
                              const int* __restrict__ batch,
                              int* __restrict__ src, int* __restrict__ dst) {
  long long i = (long long)blockIdx.x * 256 + threadIdx.x;
  if (i >= ETOT) return;
  int s, d;
  if (i < E_)               { s = edge_index[i]; d = edge_index[E_ + i]; }
  else if (i < E_ + N_)     { long long j = i - E_;        s = (int)j; d = batch[j] + (int)N_; }
  else if (i < E_ + 2*N_)   { long long j = i - (E_ + N_); s = batch[j] + (int)N_; d = (int)j; }
  else                      { long long j = i - (E_ + 2*N_); s = (int)j; d = (int)j; }
  src[i] = s; dst[i] = d;
}

// rows [0, E+2N): fill ea (bf16) and accumulate per-dst sums for loop_attr
__global__ void k_ea_sum(const float* __restrict__ edge_attr,
                         const int* __restrict__ dst,
                         bf16* __restrict__ ea_bf,
                         float* __restrict__ sumN, float* __restrict__ cntN) {
  long long i = (long long)blockIdx.x * 256 + threadIdx.x;
  long long n = (E_ + 2*N_) * 64;
  if (i >= n) return;
  long long e = i >> 6; int f = (int)(i & 63);
  float v = (e < E_) ? edge_attr[i] : (f == 0 ? 1.f : 0.f);
  ea_bf[i] = (bf16)v;
  int d = dst[e];
  if (v != 0.f) atomicAdd(&sumN[(long long)d * 64 + f], v);
  if (f == 0)   atomicAdd(&cntN[d], 1.f);
}

__global__ void k_loop_attr(const float* __restrict__ sumN, const float* __restrict__ cntN,
                            bf16* __restrict__ ea_bf) {
  long long i = (long long)blockIdx.x * 256 + threadIdx.x;
  if (i >= NTOT * 64) return;
  long long nn = i >> 6;
  float c = fmaxf(cntN[nn], 1.f);
  ea_bf[(E_ + 2*N_) * 64 + i] = (bf16)(sumN[i] / c);
}

__global__ void k_cntb(const int* __restrict__ batch, float* __restrict__ cntb) {
  long long i = (long long)blockIdx.x * 256 + threadIdx.x;
  if (i < N_) atomicAdd(&cntb[batch[i]], 1.f);
}

__global__ void k_init_h(const float* __restrict__ x, const float* __restrict__ cond,
                         float* __restrict__ h) {
  long long i = (long long)blockIdx.x * 256 + threadIdx.x;
  if (i >= NTOT * 64) return;
  long long nn = i >> 6;
  h[i] = (nn < N_) ? x[i] : cond[i - N_ * 64];
}

// ------------------------------- graph layernorm ---------------------------
// scalar mean/var per graph over all nodes+features of that graph

__device__ inline int seg_of(long long node, const int* batch) {
  return (node < N_) ? batch[node] : (int)(node - N_);
}

__global__ void k_seg_sum(const float* __restrict__ h, const int* __restrict__ batch,
                          float* __restrict__ gsum) {
  int lane = threadIdx.x & 31;
  long long node = (long long)blockIdx.x * 8 + (threadIdx.x >> 5);
  if (node >= NTOT) return;
  const float* row = h + node * 64;
  float s = row[lane] + row[lane + 32];
  #pragma unroll
  for (int m = 16; m; m >>= 1) s += __shfl_xor(s, m, 32);
  if (lane == 0) atomicAdd(&gsum[seg_of(node, batch)], s);
}

__global__ void k_ln_mean(float* __restrict__ gmean, const float* __restrict__ gsum,
                          const float* __restrict__ cntb) {
  int g = blockIdx.x * 256 + threadIdx.x;
  if (g < G_) gmean[g] = gsum[g] / ((cntb[g] + 1.f) * 64.f);
}

__global__ void k_seg_var(const float* __restrict__ h, const int* __restrict__ batch,
                          const float* __restrict__ gmean, float* __restrict__ gvar) {
  int lane = threadIdx.x & 31;
  long long node = (long long)blockIdx.x * 8 + (threadIdx.x >> 5);
  if (node >= NTOT) return;
  float m = gmean[seg_of(node, batch)];
  const float* row = h + node * 64;
  float d0 = row[lane] - m, d1 = row[lane + 32] - m;
  float s = d0 * d0 + d1 * d1;
  #pragma unroll
  for (int mm = 16; mm; mm >>= 1) s += __shfl_xor(s, mm, 32);
  if (lane == 0) atomicAdd(&gvar[seg_of(node, batch)], s);
}

__global__ void k_ln_rstd(float* __restrict__ grstd, const float* __restrict__ gvar,
                          const float* __restrict__ cntb) {
  int g = blockIdx.x * 256 + threadIdx.x;
  if (g < G_) grstd[g] = rsqrtf(gvar[g] / ((cntb[g] + 1.f) * 64.f) + 1e-5f);
}

__global__ void k_ln_apply(const float* __restrict__ hin, const int* __restrict__ batch,
                           const float* __restrict__ gmean, const float* __restrict__ grstd,
                           float* __restrict__ hout, bf16* __restrict__ bfout, int ldbf) {
  long long i = (long long)blockIdx.x * 256 + threadIdx.x;
  if (i >= NTOT * 64) return;
  long long nn = i >> 6; int f = (int)(i & 63);
  int g = seg_of(nn, batch);
  float v = (hin[i] - gmean[g]) * grstd[g];
  hout[i] = v;
  if (bfout) bfout[nn * ldbf + f] = (bf16)v;
}

// ------------------------------- message passing ---------------------------

__global__ void k_msg(const float* __restrict__ hn, const bf16* __restrict__ ea_bf,
                      const int* __restrict__ src, const int* __restrict__ dst,
                      float* __restrict__ agg) {
  long long i = (long long)blockIdx.x * 256 + threadIdx.x;
  if (i >= ETOT * 64) return;
  long long e = i >> 6; int f = (int)(i & 63);
  float v = hn[(long long)src[e] * 64 + f] + (float)ea_bf[i];
  v = (v > 0.f ? v : 0.f) + 1e-7f;
  atomicAdd(&agg[(long long)dst[e] * 64 + f], v);
}

// ------------------------------- WMMA GEMM ---------------------------------
// C[M x Nn] = act( beta*C + A[M x K](bf16) @ W[K x Nn](bf16) + bias )
// Block 256 thr (8 waves), 64x64 tile; wave -> 16x32 strip (2 WMMA accums).
// B slice staged transposed in LDS once (padded rows, bank-conflict-free);
// A fragments loaded straight from global; K loop is barrier- and
// divergence-free (A row clamped; OOB rows computed but never stored).

__global__ __launch_bounds__(256) void k_gemm_bf16(
    const bf16* __restrict__ A, long long M, int K,
    const bf16* __restrict__ W, int Nn,
    const float* __restrict__ bias, float beta, int act,
    float* __restrict__ C, int ldc,
    bf16* __restrict__ Cbf, int ldcbf)
{
  __shared__ bf16 Bs[64][264];             // [n][k], K <= 256; 528B rows
  const long long rowBase = (long long)blockIdx.y * 64;
  const int nBase = blockIdx.x * 64;
  const int t    = threadIdx.x;
  const int lane = t & 31, wid = t >> 5;
  const int wm   = (wid >> 1) * 16;        // 0,16,32,48
  const int wn   = (wid & 1) * 32;         // 0,32
  const int half = lane >> 4, nl = lane & 15;

  // stage full B slice transposed: Bs[n][k] = W[k][nBase+n]
  for (int k0 = 0; k0 < K; k0 += 32) {
    int kk = t >> 3, sg = t & 7;           // kk 0..31, sg 0..7
    const bf16* wrow = W + (long long)(k0 + kk) * Nn + nBase + sg * 8;
    #pragma unroll
    for (int j = 0; j < 8; ++j) Bs[sg * 8 + j][k0 + kk] = wrow[j];
  }
  __syncthreads();

  // A row owned by this lane; clamp (OOB rows computed, never stored)
  long long r = rowBase + wm + nl;
  if (r >= M) r = M - 1;
  const bf16* arow = A + r * (long long)K;
  const unsigned* br0 = reinterpret_cast<const unsigned*>(&Bs[wn + nl][0]);
  const unsigned* br1 = reinterpret_cast<const unsigned*>(&Bs[wn + 16 + nl][0]);

  v8f acc0 = {}; v8f acc1 = {};
  for (int k0 = 0; k0 < K; k0 += 32) {
    union Frag { v16bf v; unsigned u[8]; uint4 q[2]; };
    Frag a, b0, b1;
    // per-ISA A layout: lane half selects K octets {h*8..h*8+8, 16+h*8..}
    a.q[0] = *reinterpret_cast<const uint4*>(arow + k0 + half * 8);
    a.q[1] = *reinterpret_cast<const uint4*>(arow + k0 + 16 + half * 8);
    if (k0 + 32 < K) __builtin_prefetch(arow + k0 + 32, 0, 1);
    // B fragment: K octets {h*8..h*8+8, 16+h*8..}, 16B-aligned -> two ds_load_b128
    const int kb = (k0 >> 1) + half * 4;   // uint index within Bs row
    b0.q[0] = *reinterpret_cast<const uint4*>(br0 + kb);
    b0.q[1] = *reinterpret_cast<const uint4*>(br0 + kb + 8);
    b1.q[0] = *reinterpret_cast<const uint4*>(br1 + kb);
    b1.q[1] = *reinterpret_cast<const uint4*>(br1 + kb + 8);
    acc0 = __builtin_amdgcn_wmma_f32_16x16x32_bf16(false, a.v, false, b0.v, (short)0, acc0, false, false);
    acc1 = __builtin_amdgcn_wmma_f32_16x16x32_bf16(false, a.v, false, b1.v, (short)0, acc1, false, false);
  }

  const int gc0 = nBase + wn + nl;
  const int gc1 = gc0 + 16;
  const float bv0 = bias ? bias[gc0] : 0.f;
  const float bv1 = bias ? bias[gc1] : 0.f;
  #pragma unroll
  for (int j = 0; j < 8; ++j) {
    long long rr = rowBase + wm + half * 8 + j;
    if (rr >= M) continue;
    float o0 = acc0[j] + bv0;
    float o1 = acc1[j] + bv1;
    if (beta != 0.f && C) { o0 += beta * C[rr * ldc + gc0]; o1 += beta * C[rr * ldc + gc1]; }
    if (act == 1) { o0 = o0 > 0.f ? o0 : 0.01f * o0; o1 = o1 > 0.f ? o1 : 0.01f * o1; }
    if (C)   { C[rr * ldc + gc0] = o0;          C[rr * ldc + gc1] = o1; }
    if (Cbf) { Cbf[rr * ldcbf + gc0] = (bf16)o0; Cbf[rr * ldcbf + gc1] = (bf16)o1; }
  }
}

// ------------------------------- attention ---------------------------------

__global__ void k_alpha(const float* __restrict__ q, const float* __restrict__ kbuf,
                        const bf16* __restrict__ ee, const int* __restrict__ src,
                        const int* __restrict__ dst, float* __restrict__ alpha,
                        unsigned* __restrict__ amaxk) {
  long long e = (long long)blockIdx.x * 8 + (threadIdx.x >> 5);
  if (e >= ETOT) return;
  int lane = threadIdx.x & 31;
  long long s = src[e], d = dst[e];
  const float* qd = q    + d * 128;
  const float* ks = kbuf + s * 128;
  const bf16*  ep = ee   + e * 128;
  float p = 0.f;
  #pragma unroll
  for (int j = 0; j < 4; ++j) { int f = lane * 4 + j; p += qd[f] * (ks[f] + (float)ep[f]); }
  #pragma unroll
  for (int m = 8; m; m >>= 1) p += __shfl_xor(p, m, 16);
  if ((lane & 15) == 0) {
    int hh = lane >> 4;
    float al = p * SCALE;
    alpha[e * 2 + hh] = al;
    unsigned bits = __float_as_uint(al);
    unsigned key = (bits & 0x80000000u) ? ~bits : (bits | 0x80000000u);
    atomicMax(&amaxk[d * 2 + hh], key);
  }
}

__global__ void k_ex(const float* __restrict__ alpha, const unsigned* __restrict__ amaxk,
                     const int* __restrict__ dst, float* __restrict__ exb,
                     float* __restrict__ denom) {
  long long i = (long long)blockIdx.x * 256 + threadIdx.x;
  if (i >= ETOT * 2) return;
  long long e = i >> 1; int hh = (int)(i & 1);
  long long d = dst[e];
  unsigned key = amaxk[d * 2 + hh];
  unsigned bits = (key & 0x80000000u) ? (key & 0x7fffffffu) : ~key;
  float m = __uint_as_float(bits);
  float ex = __expf(alpha[i] - m);
  exb[i] = ex;
  atomicAdd(&denom[d * 2 + hh], ex);
}

__global__ void k_attn(const float* __restrict__ vbuf, const bf16* __restrict__ ee,
                       const float* __restrict__ exb, const float* __restrict__ denom,
                       const int* __restrict__ src, const int* __restrict__ dst,
                       float* __restrict__ tout) {
  long long e = (long long)blockIdx.x * 8 + (threadIdx.x >> 5);
  if (e >= ETOT) return;
  int lane = threadIdx.x & 31;
  long long s = src[e], d = dst[e];
  int hh = lane >> 4;
  float coef = exb[e * 2 + hh] / (denom[d * 2 + hh] + 1e-16f);
  #pragma unroll
  for (int j = 0; j < 4; ++j) {
    int f = lane * 4 + j;
    float vv = (vbuf[s * 128 + f] + (float)ee[e * 128 + f]) * coef;
    atomicAdd(&tout[d * 128 + f], vv);
  }
}

// ------------------------------- outputs -----------------------------------

__global__ void k_copy_out(const float* __restrict__ h, float* __restrict__ out) {
  long long i = (long long)blockIdx.x * 256 + threadIdx.x;
  if (i < N_ * 64) out[i] = h[i];
}
__global__ void k_glob_sum(const float* __restrict__ h, const int* __restrict__ batch,
                           float* __restrict__ gsum64) {
  long long i = (long long)blockIdx.x * 256 + threadIdx.x;
  if (i >= N_ * 64) return;
  long long nn = i >> 6; int f = (int)(i & 63);
  atomicAdd(&gsum64[(long long)batch[nn] * 64 + f], h[i]);
}
__global__ void k_glob_out(const float* __restrict__ gsum64, const float* __restrict__ cntb,
                           const float* __restrict__ h, float* __restrict__ out) {
  long long i = (long long)blockIdx.x * 256 + threadIdx.x;
  if (i >= G_ * 64) return;
  long long g = i >> 6; int f = (int)(i & 63);
  out[N_ * 64 + i] = gsum64[i] / fmaxf(cntb[g], 1.f) + h[(N_ + g) * 64 + f];
}
__global__ void k_ne(const float* __restrict__ h, const int* __restrict__ nei,
                     float* __restrict__ out) {
  long long i = (long long)blockIdx.x * 256 + threadIdx.x;
  if (i >= NE_ * 64) return;
  long long r = i >> 6; int f = (int)(i & 63);
  long long a = nei[r], b = nei[NE_ + r];
  out[(N_ + G_) * 64 + i] = h[a * 64 + f] + h[b * 64 + f];
}

// ------------------------------- host --------------------------------------

extern "C" void kernel_launch(void* const* d_in, const int* in_sizes, int n_in,
                              void* d_out, int out_size, void* d_ws, size_t ws_size,
                              hipStream_t stream) {
  const float* x         = (const float*)d_in[0];
  const float* edge_attr = (const float*)d_in[1];
  const float* cond      = (const float*)d_in[2];
  const int*   edge_index= (const int*)d_in[3];
  const int*   batch     = (const int*)d_in[4];
  const int*   nei       = (const int*)d_in[5];
  const float* gen_w = (const float*)d_in[6];   const float* gen_b = (const float*)d_in[7];
  const float* tq_w  = (const float*)d_in[8];   const float* tq_b  = (const float*)d_in[9];
  const float* tk_w  = (const float*)d_in[10];  const float* tk_b  = (const float*)d_in[11];
  const float* tv_w  = (const float*)d_in[12];  const float* tv_b  = (const float*)d_in[13];
  const float* te_w  = (const float*)d_in[14];
  const float* ts_w  = (const float*)d_in[15];  const float* ts_b  = (const float*)d_in[16];
  const float* lin_w = (const float*)d_in[17];  const float* lin_b = (const float*)d_in[18];
  const float* ff1_w = (const float*)d_in[19];  const float* ff1_b = (const float*)d_in[20];
  const float* ff2_w = (const float*)d_in[21];  const float* ff2_b = (const float*)d_in[22];
  float* out = (float*)d_out;

  // workspace bump allocator (~700 MB total)
  char* p = (char*)d_ws;
  auto alloc = [&](size_t bytes) -> void* {
    void* r = (void*)p; p += (bytes + 255) & ~(size_t)255; return r;
  };
  int*   src   = (int*)  alloc(ETOT * 4);
  int*   dst   = (int*)  alloc(ETOT * 4);
  bf16*  ea_bf = (bf16*) alloc(ETOT * 64 * 2);
  bf16*  ee_bf = (bf16*) alloc(ETOT * 128 * 2);
  float* alpha = (float*)alloc(ETOT * 2 * 4);
  float* exb   = (float*)alloc(ETOT * 2 * 4);
  float* cntN  = (float*)alloc(NTOT * 4);
  float* sumN  = (float*)alloc(NTOT * 64 * 4);
  float* cntb  = (float*)alloc(G_ * 4);
  float* h     = (float*)alloc(NTOT * 64 * 4);
  float* hn    = (float*)alloc(NTOT * 64 * 4);
  float* agg   = (float*)alloc(NTOT * 64 * 4);
  float* h2    = (float*)alloc(NTOT * 64 * 4);
  float* qb    = (float*)alloc(NTOT * 128 * 4);
  float* kb    = (float*)alloc(NTOT * 128 * 4);
  float* vb    = (float*)alloc(NTOT * 128 * 4);
  float* tout  = (float*)alloc(NTOT * 128 * 4);
  bf16*  xcat_bf = (bf16*)alloc(NTOT * 128 * 2);
  bf16*  genA_bf = (bf16*)alloc(NTOT * 64 * 2);   // reused as h2n_bf
  bf16*  tout_bf = (bf16*)alloc(NTOT * 128 * 2);
  bf16*  ff_bf   = (bf16*)alloc(NTOT * 256 * 2);
  unsigned* amaxk = (unsigned*)alloc(NTOT * 2 * 4);
  float* denom = (float*)alloc(NTOT * 2 * 4);
  float* gsum  = (float*)alloc(G_ * 4);
  float* gvar  = (float*)alloc(G_ * 4);
  float* gmean = (float*)alloc(G_ * 4);
  float* grstd = (float*)alloc(G_ * 4);
  float* gsum64= (float*)alloc(G_ * 64 * 4);
  bf16* genw_bf = (bf16*)alloc((size_t)LAYERS * 64 * 64 * 2);
  bf16* tqw_bf  = (bf16*)alloc((size_t)LAYERS * 128 * 128 * 2);
  bf16* tkw_bf  = (bf16*)alloc((size_t)LAYERS * 128 * 128 * 2);
  bf16* tvw_bf  = (bf16*)alloc((size_t)LAYERS * 128 * 128 * 2);
  bf16* tsw_bf  = (bf16*)alloc((size_t)LAYERS * 128 * 128 * 2);
  bf16* tew_bf  = (bf16*)alloc((size_t)LAYERS * 64 * 128 * 2);
  bf16* linw_bf = (bf16*)alloc((size_t)LAYERS * 128 * 64 * 2);
  bf16* ff1w_bf = (bf16*)alloc((size_t)LAYERS * 64 * 256 * 2);
  bf16* ff2w_bf = (bf16*)alloc((size_t)LAYERS * 256 * 64 * 2);

  auto fillf = [&](float* ptr, float v, long long n) {
    k_fill_f32<<<cdiv(n, 256), 256, 0, stream>>>(ptr, v, n);
  };
  auto fillu = [&](unsigned* ptr, unsigned v, long long n) {
    k_fill_u32<<<cdiv(n, 256), 256, 0, stream>>>(ptr, v, n);
  };
  auto cvt = [&](const float* s, bf16* d, long long n) {
    k_cvt_bf16<<<cdiv(n, 256), 256, 0, stream>>>(s, d, n);
  };
  auto gemm = [&](const bf16* A, long long M, int K, const bf16* W, int Nn,
                  const float* bias, float beta, int act,
                  float* C, int ldc, bf16* Cbf, int ldcbf) {
    dim3 grid(Nn / 64, (unsigned)((M + 63) / 64));
    k_gemm_bf16<<<grid, 256, 0, stream>>>(A, M, K, W, Nn, bias, beta, act, C, ldc, Cbf, ldcbf);
  };
  auto graph_ln = [&](const float* in, float* outF, bf16* outBf, int ldbf) {
    fillf(gsum, 0.f, G_);
    k_seg_sum<<<cdiv(NTOT, 8), 256, 0, stream>>>(in, batch, gsum);
    k_ln_mean<<<cdiv(G_, 256), 256, 0, stream>>>(gmean, gsum, cntb);
    fillf(gvar, 0.f, G_);
    k_seg_var<<<cdiv(NTOT, 8), 256, 0, stream>>>(in, batch, gmean, gvar);
    k_ln_rstd<<<cdiv(G_, 256), 256, 0, stream>>>(grstd, gvar, cntb);
    k_ln_apply<<<cdiv(NTOT * 64, 256), 256, 0, stream>>>(in, batch, gmean, grstd, outF, outBf, ldbf);
  };

  // ---- setup ----
  cvt(gen_w, genw_bf, (long long)LAYERS * 64 * 64);
  cvt(tq_w,  tqw_bf,  (long long)LAYERS * 128 * 128);
  cvt(tk_w,  tkw_bf,  (long long)LAYERS * 128 * 128);
  cvt(tv_w,  tvw_bf,  (long long)LAYERS * 128 * 128);
  cvt(ts_w,  tsw_bf,  (long long)LAYERS * 128 * 128);
  cvt(te_w,  tew_bf,  (long long)LAYERS * 64 * 128);
  cvt(lin_w, linw_bf, (long long)LAYERS * 128 * 64);
  cvt(ff1_w, ff1w_bf, (long long)LAYERS * 64 * 256);
  cvt(ff2_w, ff2w_bf, (long long)LAYERS * 256 * 64);

  k_build_edges<<<cdiv(ETOT, 256), 256, 0, stream>>>(edge_index, batch, src, dst);
  fillf(cntN, 0.f, NTOT);
  fillf(sumN, 0.f, NTOT * 64);
  fillf(cntb, 0.f, G_);
  k_ea_sum<<<cdiv((E_ + 2*N_) * 64, 256), 256, 0, stream>>>(edge_attr, dst, ea_bf, sumN, cntN);
  k_loop_attr<<<cdiv(NTOT * 64, 256), 256, 0, stream>>>(sumN, cntN, ea_bf);
  k_cntb<<<cdiv(N_, 256), 256, 0, stream>>>(batch, cntb);
  k_init_h<<<cdiv(NTOT * 64, 256), 256, 0, stream>>>(x, cond, h);

  // ---- layers ----
  for (int l = 0; l < LAYERS; ++l) {
    // graph LN -> hn (f32) and xcat[:, :64] (bf16)
    graph_ln(h, hn, xcat_bf, 128);

    // message passing: agg = segsum(relu(hn[src] + ea) + 1e-7)
    fillf(agg, 0.f, NTOT * 64);
    k_msg<<<cdiv(ETOT * 64, 256), 256, 0, stream>>>(hn, ea_bf, src, dst, agg);

    // gen_out = (agg + hn) @ gen_w + gen_b  -> xcat[:, 64:128] (bf16)
    k_add_cvt<<<cdiv(NTOT * 64, 256), 256, 0, stream>>>(agg, hn, genA_bf, NTOT * 64);
    gemm(genA_bf, NTOT, 64, genw_bf + (size_t)l * 64 * 64, 64,
         gen_b + l * 64, 0.f, 0, nullptr, 0, xcat_bf + 64, 128);

    // q, k, v (f32)
    gemm(xcat_bf, NTOT, 128, tqw_bf + (size_t)l * 128 * 128, 128, tq_b + l * 128, 0.f, 0, qb, 128, nullptr, 0);
    gemm(xcat_bf, NTOT, 128, tkw_bf + (size_t)l * 128 * 128, 128, tk_b + l * 128, 0.f, 0, kb, 128, nullptr, 0);
    gemm(xcat_bf, NTOT, 128, tvw_bf + (size_t)l * 128 * 128, 128, tv_b + l * 128, 0.f, 0, vb, 128, nullptr, 0);

    // ee = ea @ te_w  (big edge GEMM, bf16 out)
    gemm(ea_bf, ETOT, 64, tew_bf + (size_t)l * 64 * 128, 128, nullptr, 0.f, 0, nullptr, 0, ee_bf, 128);

    // attention: alpha -> segment max -> softmax -> weighted scatter
    fillu(amaxk, 0u, NTOT * 2);
    k_alpha<<<cdiv(ETOT, 8), 256, 0, stream>>>(qb, kb, ee_bf, src, dst, alpha, amaxk);
    fillf(denom, 0.f, NTOT * 2);
    k_ex<<<cdiv(ETOT * 2, 256), 256, 0, stream>>>(alpha, amaxk, dst, exb, denom);
    fillf(tout, 0.f, NTOT * 128);
    k_attn<<<cdiv(ETOT, 8), 256, 0, stream>>>(vb, ee_bf, exb, denom, src, dst, tout);

    // tout += xcat @ ts_w + ts_b ; emit bf16 copy
    gemm(xcat_bf, NTOT, 128, tsw_bf + (size_t)l * 128 * 128, 128, ts_b + l * 128,
         1.f, 0, tout, 128, tout_bf, 128);

    // h2 = tout @ lin_w + lin_b ; graph LN -> hn (reuse) + genA_bf (bf16)
    gemm(tout_bf, NTOT, 128, linw_bf + (size_t)l * 128 * 64, 64, lin_b + l * 64,
         0.f, 0, h2, 64, nullptr, 0);
    graph_ln(h2, hn, genA_bf, 64);

    // FFN with residual: h += leaky_relu(h2n @ ff1 + b) @ ff2 + b
    gemm(genA_bf, NTOT, 64, ff1w_bf + (size_t)l * 64 * 256, 256, ff1_b + l * 256,
         0.f, 1, nullptr, 0, ff_bf, 256);
    gemm(ff_bf, NTOT, 256, ff2w_bf + (size_t)l * 256 * 64, 64, ff2_b + l * 64,
         1.f, 0, h, 64, nullptr, 0);
  }

  // ---- outputs ----
  k_copy_out<<<cdiv(N_ * 64, 256), 256, 0, stream>>>(h, out);
  fillf(gsum64, 0.f, G_ * 64);
  k_glob_sum<<<cdiv(N_ * 64, 256), 256, 0, stream>>>(h, batch, gsum64);
  k_glob_out<<<cdiv(G_ * 64, 256), 256, 0, stream>>>(gsum64, cntb, h, out);
  k_ne<<<cdiv(NE_ * 64, 256), 256, 0, stream>>>(h, nei, out);
}
